// TripleCrossAttention_25288767439302
// MI455X (gfx1250) — matrix operand order repaired and checked by reading, not compiled, not run on hardware
//
#include <hip/hip_runtime.h>

#define BATCH 4
#define CH    256
#define NPIX  4096
#define CQKD  32

typedef __attribute__((ext_vector_type(16))) _Float16 v16h;
typedef __attribute__((ext_vector_type(8)))  _Float16 v8h;
typedef __attribute__((ext_vector_type(2)))  _Float16 v2h;
typedef __attribute__((ext_vector_type(8)))  float    v8f;
typedef __attribute__((ext_vector_type(4)))  float    v4f;

static __device__ __forceinline__ v16h frag_from2(const _Float16* p0, const _Float16* p1) {
  v8h a = *(const v8h*)p0;
  v8h b = *(const v8h*)p1;
  v16h r;
#pragma unroll
  for (int i = 0; i < 8; ++i) { r[i] = a[i]; r[i + 8] = b[i]; }
  return r;
}

static __device__ __forceinline__ v8f wmma16(v16h a, v16h b, v8f c) {
  // 8 args: (neg_a, A, neg_b, B, c_mod, C, reuse_a, reuse_b)
  return __builtin_amdgcn_wmma_f32_16x16x32_f16(false, a, false, b, (short)0, c, false, false);
}

static __device__ __forceinline__ v2h pack_f16(float a, float b) {
#if __has_builtin(__builtin_amdgcn_cvt_pkrtz)
  typedef __attribute__((ext_vector_type(2))) __fp16 v2fp16;
  v2fp16 t = __builtin_amdgcn_cvt_pkrtz(a, b);    // single v_cvt_pk_rtz_f16_f32
  return __builtin_bit_cast(v2h, t);
#else
  v2h r; r[0] = (_Float16)a; r[1] = (_Float16)b; return r;
#endif
}

// Butterfly max-reduce across each 16-lane half (pure VALU via permlane16).
static __device__ __forceinline__ float rowmax16(float v) {
#if __has_builtin(__builtin_amdgcn_permlane16)
  union { float f; int i; } u, o;
  u.f = v;
  o.i = __builtin_amdgcn_permlane16(u.i, u.i, 0x67452301u, 0xEFCDAB89u, false, false); // xor 1
  u.f = fmaxf(u.f, o.f);
  o.i = __builtin_amdgcn_permlane16(u.i, u.i, 0x54761032u, 0xDCFE98BAu, false, false); // xor 2
  u.f = fmaxf(u.f, o.f);
  o.i = __builtin_amdgcn_permlane16(u.i, u.i, 0x32107654u, 0xBA98FEDCu, false, false); // xor 4
  u.f = fmaxf(u.f, o.f);
  o.i = __builtin_amdgcn_permlane16(u.i, u.i, 0xFEDCBA98u, 0x76543210u, false, false); // xor 8
  u.f = fmaxf(u.f, o.f);
  return u.f;
#else
  v = fmaxf(v, __shfl_xor(v, 1));
  v = fmaxf(v, __shfl_xor(v, 2));
  v = fmaxf(v, __shfl_xor(v, 4));
  v = fmaxf(v, __shfl_xor(v, 8));
  return v;
#endif
}

// ---------------------------------------------------------------------------
// 1x1-conv projection: dst(o,n) = sum_c w[o][c] * src[b][c][n] + bias[o]
// mode==0: dst layout [B][N][O] f16 (Q / K fragment-friendly)
// mode==1: dst layout [B][O][N] f16 (V transposed: AV B-fragments contiguous)
// ---------------------------------------------------------------------------
__global__ __launch_bounds__(128, 2)
void proj_kernel(const float* __restrict__ src, const float* __restrict__ w,
                 const float* __restrict__ bias, _Float16* __restrict__ dst,
                 int transposed) {
  const int n  = blockIdx.x * 128 + threadIdx.x;
  const int o0 = blockIdx.y * 8;
  const int b  = blockIdx.z;
  const int O  = gridDim.y * 8;

  const float* sp = src + (size_t)b * CH * NPIX + n;   // column, stride NPIX
  const float* wp = w + (size_t)o0 * CH;

  float acc[8];
#pragma unroll
  for (int j = 0; j < 8; ++j) acc[j] = bias[o0 + j];

  for (int c = 0; c < CH; ++c) {
    float xv = sp[(size_t)c * NPIX];                   // coalesced across lanes
#pragma unroll
    for (int j = 0; j < 8; ++j) acc[j] = fmaf(wp[(size_t)j * CH + c], xv, acc[j]);
  }

  if (transposed) {
#pragma unroll
    for (int j = 0; j < 8; ++j)
      dst[((size_t)b * O + o0 + j) * NPIX + n] = (_Float16)acc[j];
  } else {
    v8h o;
#pragma unroll
    for (int j = 0; j < 8; ++j) o[j] = (_Float16)acc[j];
    *(v8h*)(dst + ((size_t)b * NPIX + n) * (size_t)O + o0) = o;
  }
}

// ---------------------------------------------------------------------------
// Two-pass flash cross-attention (per wave: 16 query rows, 256 value channels).
// S-tile 0 covers EVEN m-columns, tile 1 ODD -> a lane's two P values land in
// adjacent LDS slots (one packed b32 store), while the P A-fragment read and
// the V^T B-fragment mapping (A col k <-> m0+k) stay contiguous.
// ---------------------------------------------------------------------------
__global__ __launch_bounds__(128, 1)
void attn_kernel(const _Float16* __restrict__ Qh,   // [B][N][32]
                 const _Float16* __restrict__ Kh,   // [B][N][32]
                 const _Float16* __restrict__ Vt,   // [B][256][N]
                 const float* __restrict__ gamma,
                 const float* __restrict__ resid,   // [B][C][N]
                 float* __restrict__ outp) {        // [B][C][N]
  __shared__ _Float16 plds[4][16 * 32];             // per-wave P staging (1KB each)

  const int wave = threadIdx.x >> 5;
  const int lane = threadIdx.x & 31;
  const int lh   = lane & 15;
  const int hi   = lane >> 4;                       // half-wave select
  const int b    = blockIdx.y;
  const int q0   = blockIdx.x * 64 + wave * 16;

  // Q A-fragment (16 rows x 32): lane<16 -> row lh cols {0..15},
  // lane>=16 -> row lh cols {16..31}.
  const _Float16* qp = Qh + ((size_t)b * NPIX + q0 + lh) * CQKD + hi * 16;
  const v16h qa = frag_from2(qp, qp + 8);

  const _Float16* kbase = Kh + (size_t)b * NPIX * CQKD;
  const _Float16* vbase = Vt + (size_t)b * CH * (size_t)NPIX;
  _Float16* pw = plds[wave];
  const int prow0 = hi * 8;                         // C/D rows this lane owns

  // ---------------- pass 1: row max ----------------
  float rm[8];
#pragma unroll
  for (int r = 0; r < 8; ++r) rm[r] = -3.0e38f;

  for (int m0 = 0; m0 < NPIX; m0 += 32) {
    // tile0 = even columns (m0+2j), tile1 = odd columns (m0+2j+1)
    const _Float16* kp = kbase + (size_t)(m0 + 2 * lh) * CQKD + hi * 16;
    v16h kb0 = frag_from2(kp, kp + 8);
    v16h kb1 = frag_from2(kp + CQKD, kp + CQKD + 8);
    v8f zero = {};
    v8f s0 = wmma16(qa, kb0, zero);
    v8f s1 = wmma16(qa, kb1, zero);
#pragma unroll
    for (int r = 0; r < 8; ++r) rm[r] = fmaxf(rm[r], fmaxf(s0[r], s1[r]));
  }
#pragma unroll
  for (int r = 0; r < 8; ++r) rm[r] = rowmax16(rm[r]);  // finalize across lanes

  // ---------------- pass 2: unnormalized AV + row sums ----------------
  v8f acc[16];
#pragma unroll
  for (int t = 0; t < 16; ++t) acc[t] = {};
  v8f accl = {};                                    // row sums via ones-WMMA

  v16h ones;
#pragma unroll
  for (int i = 0; i < 16; ++i) ones[i] = (_Float16)1.0f;

  for (int m0 = 0; m0 < NPIX; m0 += 32) {
    const _Float16* kp = kbase + (size_t)(m0 + 2 * lh) * CQKD + hi * 16;
    v16h kb0 = frag_from2(kp, kp + 8);              // even m columns
    v16h kb1 = frag_from2(kp + CQKD, kp + CQKD + 8);// odd m columns
    v8f zero = {};
    v8f s0 = wmma16(qa, kb0, zero);
    v8f s1 = wmma16(qa, kb1, zero);

    // P = exp(S - rowmax); pack (even,odd) pair -> one b32 LDS store.
#pragma unroll
    for (int r = 0; r < 8; ++r) {
      float p0 = __expf(s0[r] - rm[r]);             // col m0+2*lh   -> slot 2*lh
      float p1 = __expf(s1[r] - rm[r]);             // col m0+2*lh+1 -> slot 2*lh+1
      *(v2h*)(pw + (prow0 + r) * 32 + 2 * lh) = pack_f16(p0, p1);
    }
    // Same-wave LDS store->load; DS ops are in-order, wait for the stores.
    asm volatile("s_wait_dscnt 0x0" ::: "memory");
    const _Float16* pp = pw + lh * 32 + hi * 16;
    v16h pa = frag_from2(pp, pp + 8);               // P as 16x32 A-fragment

    accl = wmma16(pa, ones, accl);                  // row sums (all cols equal)
#pragma unroll
    for (int t = 0; t < 16; ++t) {                  // out(16q x 16c) += P x V^T
      const _Float16* vp = vbase + (size_t)(t * 16 + lh) * NPIX + m0 + hi * 16;
      v16h vb = frag_from2(vp, vp + 8);
      acc[t] = wmma16(pa, vb, acc[t]);
    }
  }

  // ---------------- epilogue: vectorized (8 consecutive n per lane) --------
  const float g = gamma[0];
  float inv[8];
#pragma unroll
  for (int r = 0; r < 8; ++r) inv[r] = 1.0f / accl[r];
#pragma unroll
  for (int t = 0; t < 16; ++t) {
    const int c = t * 16 + lh;
    const size_t base = ((size_t)b * CH + c) * (size_t)NPIX + q0 + prow0;
    v4f x0 = *(const v4f*)(resid + base);
    v4f x1 = *(const v4f*)(resid + base + 4);
    v4f o0, o1;
#pragma unroll
    for (int r = 0; r < 4; ++r) {
      o0[r] = g * (acc[t][r] * inv[r]) + x0[r];
      o1[r] = g * (acc[t][r + 4] * inv[r + 4]) + x1[r];
    }
    *(v4f*)(outp + base) = o0;
    *(v4f*)(outp + base + 4) = o1;
  }
}

// ---------------------------------------------------------------------------
extern "C" void kernel_launch(void* const* d_in, const int* in_sizes, int n_in,
                              void* d_out, int out_size, void* d_ws, size_t ws_size,
                              hipStream_t stream) {
  const float* x     = (const float*)d_in[0];
  const float* y     = (const float*)d_in[1];
  const float* z     = (const float*)d_in[2];
  const float* q1_w  = (const float*)d_in[3];
  const float* q1_b  = (const float*)d_in[4];
  const float* k1_w  = (const float*)d_in[5];
  const float* k1_b  = (const float*)d_in[6];
  const float* v1_w  = (const float*)d_in[7];
  const float* v1_b  = (const float*)d_in[8];
  const float* g1    = (const float*)d_in[9];
  const float* q2_w  = (const float*)d_in[10];
  const float* q2_b  = (const float*)d_in[11];
  const float* k2_w  = (const float*)d_in[12];
  const float* k2_b  = (const float*)d_in[13];
  const float* v2_w  = (const float*)d_in[14];
  const float* v2_b  = (const float*)d_in[15];
  const float* g2    = (const float*)d_in[16];

  float* out = (float*)d_out;

  // Workspace: Qh(1MB) | Kh(1MB) | Vt(8MB)
  char* ws = (char*)d_ws;
  _Float16* Qh = (_Float16*)(ws);
  _Float16* Kh = (_Float16*)(ws + (size_t)(1 << 20));
  _Float16* Vt = (_Float16*)(ws + (size_t)(2 << 20));

  const dim3 pb(128);
  const dim3 pgQK(NPIX / 128, CQKD / 8, BATCH);
  const dim3 pgV(NPIX / 128, CH / 8, BATCH);
  const dim3 ag(NPIX / 64, BATCH);
  const dim3 ab(128);

  // ---- layer 1: cross_attn(x, y) -> out (residual x) ----
  proj_kernel<<<pgQK, pb, 0, stream>>>(x, q1_w, q1_b, Qh, 0);
  proj_kernel<<<pgQK, pb, 0, stream>>>(y, k1_w, k1_b, Kh, 0);
  proj_kernel<<<pgV,  pb, 0, stream>>>(y, v1_w, v1_b, Vt, 1);
  attn_kernel<<<ag, ab, 0, stream>>>(Qh, Kh, Vt, g1, x, out);

  // ---- layer 2: cross_attn(out, z) -> out (residual out, element-in-place) ----
  proj_kernel<<<pgQK, pb, 0, stream>>>(out, q2_w, q2_b, Qh, 0);
  proj_kernel<<<pgQK, pb, 0, stream>>>(z,   k2_w, k2_b, Kh, 0);
  proj_kernel<<<pgV,  pb, 0, stream>>>(z,   v2_w, v2_b, Vt, 1);
  attn_kernel<<<ag, ab, 0, stream>>>(Qh, Kh, Vt, g2, out, out);
}
